// EpisodicMemoryModule_57294863728685
// MI455X (gfx1250) — matrix operand order repaired and checked by reading, not compiled, not run on hardware
//
#include <hip/hip_runtime.h>

typedef __attribute__((ext_vector_type(16))) __bf16 v16bf;
typedef __attribute__((ext_vector_type(8)))  __bf16 v8bf;
typedef __attribute__((ext_vector_type(8)))  float  v8f;

#define BB  256
#define TT  127
#define TP  128
#define HH  512
#define GG1 512
#define K4H 2048

__device__ __forceinline__ __bf16 f2bf(float f) {
    unsigned u = __builtin_bit_cast(unsigned, f);
    u += 0x7FFFu + ((u >> 16) & 1u);           // round to nearest even
    unsigned short h = (unsigned short)(u >> 16);
    return __builtin_bit_cast(__bf16, h);
}

__device__ __forceinline__ float sigm(float x) { return 1.f / (1.f + __expf(-x)); }

// A fragment (16-bit 16x32, ISA 7.12.2): lanes 0-15 hold K chunks {0..7,16..23},
// lanes 16-31 hold {8..15,24..31}. rowbase points at row (lane&15), kbase mult of 32.
__device__ __forceinline__ v16bf load_a_frag(const __bf16* rowbase, int lane, int kbase) {
    int off = kbase + ((lane >> 4) << 3);
    v8bf lo = *(const v8bf*)(rowbase + off);
    v8bf hi = *(const v8bf*)(rowbase + off + 16);
    return __builtin_shufflevector(lo, hi, 0,1,2,3,4,5,6,7,8,9,10,11,12,13,14,15);
}

// B fragment (32x16): lane n holds column n, K {0..7,16..23}; lane n+16 holds K {8..15,24..31}.
// colbase = weight row (== B column) base; kbase mult of 32. Two 16-byte loads.
__device__ __forceinline__ v16bf load_b_frag(const __bf16* colbase, int lane, int kbase) {
    int off = kbase + ((lane >> 4) << 3);
    v8bf lo = *(const v8bf*)(colbase + off);
    v8bf hi = *(const v8bf*)(colbase + off + 16);
    return __builtin_shufflevector(lo, hi, 0,1,2,3,4,5,6,7,8,9,10,11,12,13,14,15);
}

#define WMMA_BF16(A, Bm, C) \
    __builtin_amdgcn_wmma_f32_16x16x32_bf16(false, (A), false, (Bm), (short)0, (C), false, false)

// ---------------------------------------------------------------------------
// Kernel 1: f32 -> bf16 weight conversion (runs once per launch, tiny)
// ---------------------------------------------------------------------------
__global__ __launch_bounds__(256)
void cvt_weights(const float* __restrict__ g1w, const float* __restrict__ wih,
                 const float* __restrict__ whh,
                 __bf16* __restrict__ g1o, __bf16* __restrict__ wiho,
                 __bf16* __restrict__ whho) {
    const int stride = gridDim.x * blockDim.x;
    const int tid = blockIdx.x * blockDim.x + threadIdx.x;
    for (int i = tid; i < GG1 * K4H; i += stride) g1o[i] = f2bf(g1w[i]);
    for (int i = tid; i < 3 * HH * HH; i += stride) {
        wiho[i] = f2bf(wih[i]);
        whho[i] = f2bf(whh[i]);
    }
}

// ---------------------------------------------------------------------------
// Kernel 2: gate network  G[b,t] = tanh(Z @ g1_w.T + g1_b) @ g2_w.T + g2_b
// grid = B * 8 (t-tiles of 16), block = 256 (8 waves), M=16 rows = t values.
// K=2048 processed in two 1024 halves (LDS Z tile 32KB bf16).
// ---------------------------------------------------------------------------
__global__ __launch_bounds__(256)
void gate_kernel(const float* __restrict__ s_rep, const float* __restrict__ q_rep,
                 const float* __restrict__ mem,   const __bf16* __restrict__ g1_bf,
                 const float* __restrict__ g1_b,  const float* __restrict__ g2_w,
                 const float* __restrict__ g2_b,
                 float* __restrict__ outG, float* __restrict__ Gs) {
    __shared__ alignas(32) __bf16 Zb[16][1024];
    __shared__ float gsum[16];

    const int tid  = threadIdx.x;
    const int lane = tid & 31;
    const int wave = tid >> 5;
    const int b    = blockIdx.x >> 3;
    const int t0   = (blockIdx.x & 7) << 4;

    if (tid < 16) gsum[tid] = 0.f;

    const v8f vzero = {0.f,0.f,0.f,0.f,0.f,0.f,0.f,0.f};
    v8f acc[4] = {vzero, vzero, vzero, vzero};

    for (int kh = 0; kh < 2; ++kh) {
        // build Z half-tile (16 x 1024) in bf16
        for (int i = tid; i < 16 * 1024; i += 256) {
            int m  = i >> 10;
            int kl = i & 1023;
            int k  = (kh << 10) + kl;
            int h  = k & (HH - 1);
            int seg = k >> 9;
            int t  = t0 + m;
            float s  = (t < TT) ? s_rep[((size_t)b * TT + t) * HH + h] : 0.f;
            float qv = q_rep[b * HH + h];
            float mv = mem[b * HH + h];
            float z;
            if      (seg == 0) z = s * qv;
            else if (seg == 1) z = s * mv;
            else if (seg == 2) z = fabsf(s - qv);
            else               z = fabsf(s - mv);
            Zb[m][kl] = f2bf(z);
        }
        __syncthreads();

        const __bf16* arow = &Zb[lane & 15][0];
#pragma unroll
        for (int nt = 0; nt < 4; ++nt) {
            const int n0 = (wave * 4 + nt) * 16;
            const __bf16* bcol = g1_bf + (size_t)(n0 + (lane & 15)) * K4H + (kh << 10);
            for (int kb = 0; kb < 32; ++kb) {
                v16bf a  = load_a_frag(arow, lane, kb * 32);
                v16bf bm = load_b_frag(bcol, lane, kb * 32);
                acc[nt] = WMMA_BF16(a, bm, acc[nt]);
            }
        }
        __syncthreads();
    }

    // epilogue: tanh + dot with g2_w. D fragment: lane = column, VGPR v = row
    float p[8] = {0.f,0.f,0.f,0.f,0.f,0.f,0.f,0.f};
#pragma unroll
    for (int nt = 0; nt < 4; ++nt) {
        const int ncol = (wave * 4 + nt) * 16 + (lane & 15);
        const float gb = g1_b[ncol];
        const float g2 = g2_w[ncol];
#pragma unroll
        for (int v = 0; v < 8; ++v)
            p[v] += tanhf(acc[nt][v] + gb) * g2;
    }
#pragma unroll
    for (int v = 0; v < 8; ++v)
        for (int off = 8; off >= 1; off >>= 1)
            p[v] += __shfl_xor(p[v], off, 16);
    if ((lane & 15) == 0) {
#pragma unroll
        for (int v = 0; v < 8; ++v)
            atomicAdd(&gsum[v + ((lane >> 4) << 3)], p[v]);
    }
    __syncthreads();

    if (tid < 16) {
        float Gv = gsum[tid] + g2_b[0];
        int t = t0 + tid;
        outG[b * TP + t] = Gv;                 // pre-sigmoid (second output)
        Gs[b * TP + t]   = sigm(Gv);           // gate for the scan
    }
}

// ---------------------------------------------------------------------------
// Kernel 3: gated-GRU scan. grid = 16 blocks (M=16 batch rows), 512 thr = 16 waves.
// Wave w owns H-column tiles {2w, 2w+1}; 6 WMMA accumulators (ir,iz,in,hr,hz,hn).
// h kept in f32 registers (fragment mapping is step-invariant) + bf16 LDS mirror.
// ---------------------------------------------------------------------------
__global__ __launch_bounds__(512)
void gru_kernel(const float* __restrict__ s_rep, const int* __restrict__ e_lens,
                const __bf16* __restrict__ wih,  const __bf16* __restrict__ whh,
                const float* __restrict__ b_ih,  const float* __restrict__ b_hh,
                const float* __restrict__ Gs,    float* __restrict__ out_sel) {
    __shared__ alignas(32) __bf16 hb[16][HH];
    __shared__ alignas(32) __bf16 xb[16][HH];
    __shared__ float gcur[16];
    __shared__ int   elen[16];

    const int tid  = threadIdx.x;
    const int lane = tid & 31;
    const int wave = tid >> 5;
    const int b0   = blockIdx.x * 16;
    const int mrow = ((lane >> 4) << 3);       // row offset of this lane's D fragment

    for (int i = tid; i < 16 * HH; i += 512) (&hb[0][0])[i] = f2bf(0.f);
    if (tid < 16) elen[tid] = e_lens[b0 + tid];

    float hreg[2][8];
#pragma unroll
    for (int ht = 0; ht < 2; ++ht)
#pragma unroll
        for (int v = 0; v < 8; ++v) hreg[ht][v] = 0.f;

    const v8f vzero = {0.f,0.f,0.f,0.f,0.f,0.f,0.f,0.f};

    for (int t = 0; t < TP; ++t) {
        // stage x_t tile (16 x 512) as bf16 (t == 127 is the appended zero row)
        for (int i = tid; i < 16 * HH; i += 512) {
            int m = i >> 9;
            int h = i & (HH - 1);
            float xv = (t < TT) ? s_rep[((size_t)(b0 + m) * TT + t) * HH + h] : 0.f;
            (&xb[0][0])[i] = f2bf(xv);
        }
        if (tid < 16) gcur[tid] = Gs[(b0 + tid) * TP + t];
        __syncthreads();

        float hnew[2][8];
        const __bf16* xrow = &xb[lane & 15][0];
        const __bf16* hrow = &hb[lane & 15][0];
#pragma unroll
        for (int ht = 0; ht < 2; ++ht) {
            const int col = (wave * 2 + ht) * 16 + (lane & 15);
            v8f acc[6] = {vzero, vzero, vzero, vzero, vzero, vzero};
            for (int kb = 0; kb < 16; ++kb) {
                v16bf ax = load_a_frag(xrow, lane, kb * 32);
                v16bf ah = load_a_frag(hrow, lane, kb * 32);
#pragma unroll
                for (int g = 0; g < 3; ++g) {
                    const __bf16* bi = wih + (size_t)(g * HH + col) * HH;
                    const __bf16* bh = whh + (size_t)(g * HH + col) * HH;
                    v16bf fi = load_b_frag(bi, lane, kb * 32);
                    v16bf fh = load_b_frag(bh, lane, kb * 32);
                    acc[g]     = WMMA_BF16(ax, fi, acc[g]);
                    acc[3 + g] = WMMA_BF16(ah, fh, acc[3 + g]);
                }
            }
            const float bir = b_ih[col],          bhr = b_hh[col];
            const float biz = b_ih[HH + col],     bhz = b_hh[HH + col];
            const float bin = b_ih[2 * HH + col], bhn = b_hh[2 * HH + col];
#pragma unroll
            for (int v = 0; v < 8; ++v) {
                const int m = v + mrow;
                const float hold = hreg[ht][v];
                const float r  = sigm(acc[0][v] + bir + acc[3][v] + bhr);
                const float z  = sigm(acc[1][v] + biz + acc[4][v] + bhz);
                const float n  = tanhf(acc[2][v] + bin + r * (acc[5][v] + bhn));
                const float hg = (1.f - z) * n + z * hold;
                const float gg = gcur[m];
                const float h2 = gg * hg + (1.f - gg) * hold;
                hnew[ht][v] = h2;
                if (elen[m] - 1 == t)
                    out_sel[(size_t)(b0 + m) * HH + col] = h2;
            }
        }
        __syncthreads();   // all waves done reading hb for step t
#pragma unroll
        for (int ht = 0; ht < 2; ++ht) {
            const int col = (wave * 2 + ht) * 16 + (lane & 15);
#pragma unroll
            for (int v = 0; v < 8; ++v) {
                hb[v + mrow][col] = f2bf(hnew[ht][v]);
                hreg[ht][v] = hnew[ht][v];
            }
        }
        __syncthreads();   // hb updated before next step's reads
    }
}

// ---------------------------------------------------------------------------
extern "C" void kernel_launch(void* const* d_in, const int* in_sizes, int n_in,
                              void* d_out, int out_size, void* d_ws, size_t ws_size,
                              hipStream_t stream) {
    const float* s_rep  = (const float*)d_in[0];
    const float* q_rep  = (const float*)d_in[1];
    const float* memory = (const float*)d_in[2];
    const int*   e_lens = (const int*)  d_in[3];
    const float* W_ih   = (const float*)d_in[4];
    const float* W_hh   = (const float*)d_in[5];
    const float* b_ih   = (const float*)d_in[6];
    const float* b_hh   = (const float*)d_in[7];
    const float* g1_w   = (const float*)d_in[8];
    const float* g1_b   = (const float*)d_in[9];
    const float* g2_w   = (const float*)d_in[10];
    const float* g2_b   = (const float*)d_in[11];

    float* out_sel = (float*)d_out;            // (B, H)
    float* out_G   = out_sel + BB * HH;        // (B, 128) pre-sigmoid

    char* ws = (char*)d_ws;
    __bf16* g1_bf  = (__bf16*)ws;                                        // 2 MB
    __bf16* wih_bf = (__bf16*)(ws + (size_t)GG1 * K4H * 2);              // 1.5 MB
    __bf16* whh_bf = (__bf16*)(ws + (size_t)GG1 * K4H * 2 + (size_t)3 * HH * HH * 2);
    float*  Gs     = (float*) (ws + (size_t)GG1 * K4H * 2 + (size_t)6 * HH * HH * 2);

    cvt_weights<<<2048, 256, 0, stream>>>(g1_w, W_ih, W_hh, g1_bf, wih_bf, whh_bf);
    gate_kernel<<<BB * 8, 256, 0, stream>>>(s_rep, q_rep, memory, g1_bf,
                                            g1_b, g2_w, g2_b, out_G, Gs);
    gru_kernel<<<BB / 16, 512, 0, stream>>>(s_rep, e_lens, wih_bf, whh_bf,
                                            b_ih, b_hh, Gs, out_sel);
}
